// UNetKPCMLarge_50379966382789
// MI455X (gfx1250) — matrix operand order repaired and checked by reading, not compiled, not run on hardware
//
#include <hip/hip_runtime.h>
#include <hip/hip_bf16.h>
#include <stdint.h>

// ---------------------------------------------------------------------------
// MI455X (gfx1250, wave32) UNet-KPN forward, round 3.
//
// Layout strategy (derived from the CDNA5 WMMA fragment maps, ISA 7.12.2):
//  * Activations: NHWC, zero-padded spatially: [(H+2)*(W+2)][C] fp16.
//  * Weights: pre-packed fp16, tap-major: [Cout][9][Cin]  (k = tap*Cin+ci).
//  => per lane, an A fragment is one contiguous 16-half chunk
//     (k0+hlf*16) and a B fragment is one contiguous 16-half
//     chunk (k0+hlf*16) of its own pixel's channels: everything loads as
//     16-byte global_load_b128, no per-element gather, no cvt in the loop.
//  * Epilogue: 8 consecutive Cout channels per lane -> one b128 store/tile.
//  * Skip-concat: fragment channels never straddle the C1 boundary
//    (C1 % 16 == 0), so it is a single uniform pointer select.
// ---------------------------------------------------------------------------

typedef __attribute__((ext_vector_type(16))) _Float16 v16h;
typedef __attribute__((ext_vector_type(8)))  _Float16 v8h;
typedef __attribute__((ext_vector_type(8)))  float    v8f;

#define KW   35
#define K2   1225
#define KPAD 17

__device__ __forceinline__ v16h cat16(v8h lo, v8h hi) {
    return __builtin_shufflevector(lo, hi, 0, 1, 2, 3, 4, 5, 6, 7,
                                   8, 9, 10, 11, 12, 13, 14, 15);
}

// ---------------------------------------------------------------------------
// arena zero-fill (padded borders + padded channels of activations must be 0)
// ---------------------------------------------------------------------------
__global__ void fill_zero_b16(uint4* __restrict__ p, long n16) {
    long i = (long)blockIdx.x * blockDim.x + threadIdx.x;
    if (i < n16) p[i] = make_uint4(0u, 0u, 0u, 0u);
}

// ---------------------------------------------------------------------------
// conv weight repack: OIHW fp32 [Cout][CinS][3][3] -> fp16 [Cout][9][CinD]
// (tap-major K; channels zero-padded CinS -> CinD)
// ---------------------------------------------------------------------------
__global__ void repack_conv_w(const float* __restrict__ w,
                              _Float16* __restrict__ out,
                              int Cout, int CinS, int CinD) {
    int i = blockIdx.x * blockDim.x + threadIdx.x;
    int cinD9 = 9 * CinD;
    if (i >= Cout * cinD9) return;
    int o   = i / cinD9;
    int r   = i - o * cinD9;
    int tap = r / CinD;
    int ci  = r - tap * CinD;
    out[i] = (ci < CinS) ? (_Float16)w[((size_t)o * CinS + ci) * 9 + tap]
                         : (_Float16)0.0f;
}

// ---------------------------------------------------------------------------
// up-conv weight repack: fp32 [Cin][Cout][2][2] -> fp16 [4][Cout][Cin]
// ---------------------------------------------------------------------------
__global__ void repack_up_w(const float* __restrict__ w,
                            _Float16* __restrict__ out, int Cin, int Cout) {
    int i = blockIdx.x * blockDim.x + threadIdx.x;
    int per = Cout * Cin;
    if (i >= 4 * per) return;
    int sub = i / per;
    int r   = i - sub * per;
    int o   = r / Cin;
    int ci  = r - o * Cin;
    out[i] = (_Float16)w[((size_t)ci * Cout + o) * 4 + sub];
}

// ---------------------------------------------------------------------------
// head weight repack: fp32 [1225][128] -> fp16 [1232][128] (rows zero-padded)
// ---------------------------------------------------------------------------
__global__ void repack_head_w(const float* __restrict__ w,
                              _Float16* __restrict__ out) {
    int i = blockIdx.x * blockDim.x + threadIdx.x;
    if (i >= 1232 * 128) return;
    int m = i >> 7, c = i & 127;
    out[i] = (m < K2) ? (_Float16)w[m * 128 + c] : (_Float16)0.0f;
}

// ---------------------------------------------------------------------------
// noisy fp32 [3][H][W] -> fp16 NHWC padded [(H+2)(W+2)][32] (ch 3..31 zero)
// ---------------------------------------------------------------------------
__global__ void pack_noisy_kernel(const float* __restrict__ in,
                                  _Float16* __restrict__ out, int H, int W) {
    int p = blockIdx.x * blockDim.x + threadIdx.x;
    int HW = H * W;
    if (p >= HW) return;
    int y = p / W, x = p - y * W;
    size_t pix = (size_t)(y + 1) * (W + 2) + (x + 1);
    out[pix * 32 + 0] = (_Float16)in[p];
    out[pix * 32 + 1] = (_Float16)in[HW + p];
    out[pix * 32 + 2] = (_Float16)in[2 * HW + p];
}

// ---------------------------------------------------------------------------
// 2x2 max-pool, NHWC padded -> NHWC padded (coalesced over channels)
// ---------------------------------------------------------------------------
__global__ void maxpool2_h(const _Float16* __restrict__ in,
                           _Float16* __restrict__ out, int C, int H, int W) {
    int idx = blockIdx.x * blockDim.x + threadIdx.x;
    int oh = H >> 1, ow = W >> 1;
    if (idx >= C * oh * ow) return;
    int c  = idx % C;
    int pp = idx / C;
    int y = pp / ow, x = pp - y * ow;
    int Wp = W + 2, Wq = ow + 2;
    const _Float16* p = in + ((size_t)(2 * y + 1) * Wp + (2 * x + 1)) * C + c;
    float m = fmaxf(fmaxf((float)p[0], (float)p[C]),
                    fmaxf((float)p[(size_t)Wp * C], (float)p[(size_t)Wp * C + C]));
    out[((size_t)(y + 1) * Wq + (x + 1)) * C + c] = (_Float16)m;
}

// ---------------------------------------------------------------------------
// Implicit-GEMM 3x3 conv + bias + ReLU via WMMA. NHWC fp16, packed weights.
//   in1: channels [0,C1) stride C1; in2: channels [C1,Cpad) stride C2.
//   wgt: fp16 [Cout][9*Cpad] tap-major.  Cpad % 32 == 0.
// Each wave: 16 Cout x 64 pixels (4 tiles).  Grid: (H*W/64, Cout/128), 256.
// ---------------------------------------------------------------------------
__global__ __launch_bounds__(256)
void conv3x3_wmma(const _Float16* __restrict__ in1,
                  const _Float16* __restrict__ in2,
                  int C1, int C2, int Cpad,
                  const _Float16* __restrict__ wgt,
                  const float* __restrict__ bias,
                  _Float16* __restrict__ out,
                  int Cout, int H, int W) {
    const int Wp   = W + 2;
    const int Kdim = 9 * Cpad;
    const int lane = threadIdx.x & 31;
    const int wave = threadIdx.x >> 5;
    const int m16  = lane & 15;
    const int hlf  = lane >> 4;
    const int pixBase  = blockIdx.x * 64;
    const int coutBase = (blockIdx.y * 8 + wave) * 16;

    int baseT[4];                       // padded interior pixel index
#pragma unroll
    for (int t = 0; t < 4; ++t) {
        const int n  = pixBase + t * 16 + m16;
        const int py = n / W;
        const int px = n - py * W;
        baseT[t] = (py + 1) * Wp + (px + 1);
    }
    const _Float16* wrow = wgt + (size_t)(coutBase + m16) * Kdim;

    v8f acc[4] = {{}, {}, {}, {}};
    for (int tap = 0; tap < 9; ++tap) {
        const int ky = tap / 3, kx = tap - (tap / 3) * 3;
        const int soff = (ky - 1) * Wp + (kx - 1);      // tap pixel offset
        const _Float16* wtap = wrow + tap * Cpad;
        for (int c0 = 0; c0 < Cpad; c0 += 32) {
            __builtin_prefetch(wtap + c0 + 512, 0, 3);
            // A fragment: one contiguous 16-half chunk
            const _Float16* ap = wtap + c0 + hlf * 16;
            const v16h a = cat16(*(const v8h*)ap, *(const v8h*)(ap + 8));
            // B fragments: one contiguous 16-half chunk per pixel
            const int  ci    = c0 + hlf * 16;
            const bool first = ci < C1;
            const _Float16* src = first ? in1 : in2;
            const int cc = first ? ci : ci - C1;
            const int Ct = first ? C1 : C2;
            v16h bf[4];
#pragma unroll
            for (int t = 0; t < 4; ++t) {
                const _Float16* bp = src + (size_t)((baseT[t] + soff) * Ct + cc);
                bf[t] = cat16(*(const v8h*)bp, *(const v8h*)(bp + 8));
            }
#pragma unroll
            for (int t = 0; t < 4; ++t)
                acc[t] = __builtin_amdgcn_wmma_f32_16x16x32_f16(
                    false, a, false, bf[t], (short)0, acc[t], false, false);
        }
    }
    // epilogue: bias + ReLU; 8 consecutive Cout channels -> one b128 store
#pragma unroll
    for (int t = 0; t < 4; ++t) {
        v8h o;
#pragma unroll
        for (int v = 0; v < 8; ++v) {
            const int co = coutBase + 8 * hlf + v;
            o[v] = (_Float16)fmaxf(acc[t][v] + bias[co], 0.0f);
        }
        *(v8h*)(out + (size_t)baseT[t] * Cout + coutBase + 8 * hlf) = o;
    }
}

// ---------------------------------------------------------------------------
// ConvTranspose2d(k=2,s=2) as 4 sub-position 1x1 GEMMs via WMMA.
//   wgt: fp16 [4][Cout][Cin] packed; NHWC in/out. Grid: (HW/16, Cout/128, 4).
// ---------------------------------------------------------------------------
__global__ __launch_bounds__(256)
void upconv2x2_wmma(const _Float16* __restrict__ in,
                    const _Float16* __restrict__ wgt,
                    const float* __restrict__ bias,
                    _Float16* __restrict__ out,
                    int Cin, int Cout, int H, int W) {
    const int Wp = W + 2;
    const int Wq = 2 * W + 2;
    const int lane = threadIdx.x & 31;
    const int wave = threadIdx.x >> 5;
    const int m16  = lane & 15;
    const int hlf  = lane >> 4;
    const int sub  = blockIdx.z;
    const int su = sub >> 1, sv = sub & 1;
    const int coutBase = (blockIdx.y * 8 + wave) * 16;

    const int n  = blockIdx.x * 16 + m16;
    const int py = n / W;
    const int px = n - py * W;
    const size_t ibase = ((size_t)(py + 1) * Wp + (px + 1)) * Cin;

    const _Float16* wrow = wgt + ((size_t)sub * Cout + coutBase + m16) * Cin;

    v8f acc = {};
    for (int c0 = 0; c0 < Cin; c0 += 32) {
        const _Float16* ap = wrow + c0 + hlf * 16;
        const v16h a = cat16(*(const v8h*)ap, *(const v8h*)(ap + 8));
        const _Float16* bp = in + ibase + c0 + hlf * 16;
        const v16h b = cat16(*(const v8h*)bp, *(const v8h*)(bp + 8));
        acc = __builtin_amdgcn_wmma_f32_16x16x32_f16(
            false, a, false, b, (short)0, acc, false, false);
    }
    v8h o;
#pragma unroll
    for (int v = 0; v < 8; ++v) {
        const int co = coutBase + 8 * hlf + v;
        o[v] = (_Float16)(acc[v] + bias[co]);
    }
    *(v8h*)(out + ((size_t)(2 * py + su + 1) * Wq + (2 * px + sv + 1)) * Cout
            + coutBase + 8 * hlf) = o;
}

// ---------------------------------------------------------------------------
// KPN head: raw[m][p] = sum_c wk[m][c] * d1[p][c] + bk[m]
//   wk: fp16 [1232][128] (rows zero-padded); d1: NHWC [pix][128].
//   raw stays [K2][HW] fp32 so kpn_apply reads coalesce.
// Grid: (HW/16, 10), block 256 (77 M-tiles, tail waves masked).
// ---------------------------------------------------------------------------
__global__ __launch_bounds__(256)
void kpn_head_wmma(const _Float16* __restrict__ d1,
                   const _Float16* __restrict__ wk,
                   const float* __restrict__ bk,
                   float* __restrict__ raw, int M, int H, int W) {
    const int HW = H * W;
    const int Wp = W + 2;
    const int lane = threadIdx.x & 31;
    const int wave = threadIdx.x >> 5;
    const int m16  = lane & 15;
    const int hlf  = lane >> 4;
    const int mBase = (blockIdx.y * 8 + wave) * 16;
    if (mBase >= M) return;                     // wave-uniform early-out
    const int n  = blockIdx.x * 16 + m16;
    const int py = n / W;
    const int px = n - py * W;
    const size_t ibase = ((size_t)(py + 1) * Wp + (px + 1)) * 128;

    const _Float16* wrow = wk + (size_t)(mBase + m16) * 128;

    v8f acc = {};
#pragma unroll
    for (int c0 = 0; c0 < 128; c0 += 32) {
        const _Float16* ap = wrow + c0 + hlf * 16;
        const v16h a = cat16(*(const v8h*)ap, *(const v8h*)(ap + 8));
        const _Float16* bp = d1 + ibase + c0 + hlf * 16;
        const v16h b = cat16(*(const v8h*)bp, *(const v8h*)(bp + 8));
        acc = __builtin_amdgcn_wmma_f32_16x16x32_f16(
            false, a, false, b, (short)0, acc, false, false);
    }
#pragma unroll
    for (int v = 0; v < 8; ++v) {
        const int m = mBase + v + 8 * hlf;
        if (m < M) raw[(size_t)m * HW + n] = acc[v] + bk[m];
    }
}

// ---------------------------------------------------------------------------
// Fused normalization + kernel application (output == weighted patch sum).
// ---------------------------------------------------------------------------
__global__ void kpn_apply(const float* __restrict__ raw,
                          const float* __restrict__ noisy,
                          float* __restrict__ out, int H, int W) {
    const int HW = H * W;
    const int p = blockIdx.x * blockDim.x + threadIdx.x;
    if (p >= HW) return;
    const int py = p / W;
    const int px = p - py * W;

    float s = 0.0f, a0 = 0.0f, a1 = 0.0f, a2 = 0.0f;
    for (int k = 0; k < K2; ++k) {
        const float wv = raw[(size_t)k * HW + p];
        s += wv;
        const int ky = k / KW;
        const int kx = k - ky * KW;
        const int iy = py + ky - KPAD;
        const int ix = px + kx - KPAD;
        if ((unsigned)iy < (unsigned)H && (unsigned)ix < (unsigned)W) {
            const int off = iy * W + ix;
            a0 += wv * noisy[off];
            a1 += wv * noisy[HW + off];
            a2 += wv * noisy[2 * HW + off];
        }
    }
    const float inv = 1.0f / fmaxf(s, 1e-6f);
    out[p]          = a0 * inv;
    out[HW + p]     = a1 * inv;
    out[2 * HW + p] = a2 * inv;
}

// ---------------------------------------------------------------------------
// Host orchestration
// ---------------------------------------------------------------------------
extern "C" void kernel_launch(void* const* d_in, const int* in_sizes, int n_in,
                              void* d_out, int out_size, void* d_ws, size_t ws_size,
                              hipStream_t stream) {
    (void)in_sizes; (void)n_in; (void)out_size; (void)ws_size;

    const float* noisy = (const float*)d_in[0];
    auto P = [&](int i) { return (const float*)d_in[i]; };
    // 1..4 e1 | 5..8 e2 | 9..12 e3 | 13..16 e4 | 17..20 bn | 21,22 up4
    // 23..26 dec4 | 27,28 up3 | 29..32 dec3 | 33,34 up2 | 35..38 dec2
    // 39,40 up1 | 41..44 dec1 | 45 outk_w | 46 outk_b

    char* wp = (char*)d_ws;
    auto allocH = [&](size_t nelem) -> _Float16* {
        wp = (char*)(((uintptr_t)wp + 255) & ~(uintptr_t)255);
        _Float16* q = (_Float16*)wp; wp += nelem * sizeof(_Float16); return q;
    };
    auto allocF = [&](size_t nelem) -> float* {
        wp = (char*)(((uintptr_t)wp + 255) & ~(uintptr_t)255);
        float* q = (float*)wp; wp += nelem * sizeof(float); return q;
    };

    const int H = 128, W = 128, HW = H * W;
    const size_t A1 = (size_t)130 * 130;   // padded plane @128
    const size_t A2 = (size_t)66 * 66;     // @64
    const size_t A3 = (size_t)34 * 34;     // @32
    const size_t A4 = (size_t)18 * 18;     // @16
    const size_t A5 = (size_t)10 * 10;     // @8

    // ---- fp16 weight arena (fully overwritten by repack kernels) ----------
    _Float16* we11 = allocH((size_t)128 * 9 * 32);
    _Float16* we12 = allocH((size_t)128 * 9 * 128);
    _Float16* we21 = allocH((size_t)256 * 9 * 128);
    _Float16* we22 = allocH((size_t)256 * 9 * 256);
    _Float16* we31 = allocH((size_t)512 * 9 * 256);
    _Float16* we32 = allocH((size_t)512 * 9 * 512);
    _Float16* we41 = allocH((size_t)1024 * 9 * 512);
    _Float16* we42 = allocH((size_t)1024 * 9 * 1024);
    _Float16* wbn1 = allocH((size_t)2048 * 9 * 1024);
    _Float16* wbn2 = allocH((size_t)2048 * 9 * 2048);
    _Float16* wd41 = allocH((size_t)1024 * 9 * 2048);
    _Float16* wd42 = allocH((size_t)1024 * 9 * 1024);
    _Float16* wd31 = allocH((size_t)512 * 9 * 1024);
    _Float16* wd32 = allocH((size_t)512 * 9 * 512);
    _Float16* wd21 = allocH((size_t)256 * 9 * 512);
    _Float16* wd22 = allocH((size_t)256 * 9 * 256);
    _Float16* wd11 = allocH((size_t)128 * 9 * 256);
    _Float16* wd12 = allocH((size_t)128 * 9 * 128);
    _Float16* wu4  = allocH((size_t)4 * 1024 * 2048);
    _Float16* wu3  = allocH((size_t)4 * 512 * 1024);
    _Float16* wu2  = allocH((size_t)4 * 256 * 512);
    _Float16* wu1  = allocH((size_t)4 * 128 * 256);
    _Float16* whd  = allocH((size_t)1232 * 128);

    // ---- NHWC activation arena (zero-filled once: borders + pad channels) -
    char* arena0 = (char*)(((uintptr_t)wp + 255) & ~(uintptr_t)255);
    _Float16* nz  = allocH(32 * A1);
    _Float16 *t1 = allocH(128 * A1), *x1 = allocH(128 * A1);
    _Float16 *u1 = allocH(128 * A1), *d1m = allocH(128 * A1);
    _Float16 *d1 = allocH(128 * A1);
    _Float16 *p1 = allocH(128 * A2);
    _Float16 *t2 = allocH(256 * A2), *x2 = allocH(256 * A2);
    _Float16 *u2 = allocH(256 * A2), *d2m = allocH(256 * A2);
    _Float16 *d2 = allocH(256 * A2);
    _Float16 *p2 = allocH(256 * A3);
    _Float16 *t3 = allocH(512 * A3), *x3 = allocH(512 * A3);
    _Float16 *u3 = allocH(512 * A3), *d3m = allocH(512 * A3);
    _Float16 *d3 = allocH(512 * A3);
    _Float16 *p3 = allocH(512 * A4);
    _Float16 *t4 = allocH(1024 * A4), *x4 = allocH(1024 * A4);
    _Float16 *u4 = allocH(1024 * A4), *d4m = allocH(1024 * A4);
    _Float16 *d4 = allocH(1024 * A4);
    _Float16 *p4 = allocH(1024 * A5);
    _Float16 *bnm = allocH(2048 * A5), *bno = allocH(2048 * A5);
    char* arena1 = wp;
    float* raw = allocF((size_t)K2 * HW);

    // ---- prep -------------------------------------------------------------
    {
        long n16 = (long)((arena1 - arena0 + 15) / 16);
        fill_zero_b16<<<(n16 + 255) / 256, 256, 0, stream>>>((uint4*)arena0, n16);
    }
    auto rpc = [&](const float* w, _Float16* o, int cout, int cinS, int cinD) {
        int n = cout * 9 * cinD;
        repack_conv_w<<<(n + 255) / 256, 256, 0, stream>>>(w, o, cout, cinS, cinD);
    };
    auto rpu = [&](const float* w, _Float16* o, int cin, int cout) {
        int n = 4 * cout * cin;
        repack_up_w<<<(n + 255) / 256, 256, 0, stream>>>(w, o, cin, cout);
    };
    rpc(P(1),  we11, 128, 3, 32);     rpc(P(3),  we12, 128, 128, 128);
    rpc(P(5),  we21, 256, 128, 128);  rpc(P(7),  we22, 256, 256, 256);
    rpc(P(9),  we31, 512, 256, 256);  rpc(P(11), we32, 512, 512, 512);
    rpc(P(13), we41, 1024, 512, 512); rpc(P(15), we42, 1024, 1024, 1024);
    rpc(P(17), wbn1, 2048, 1024, 1024); rpc(P(19), wbn2, 2048, 2048, 2048);
    rpc(P(23), wd41, 1024, 2048, 2048); rpc(P(25), wd42, 1024, 1024, 1024);
    rpc(P(29), wd31, 512, 1024, 1024);  rpc(P(31), wd32, 512, 512, 512);
    rpc(P(35), wd21, 256, 512, 512);    rpc(P(37), wd22, 256, 256, 256);
    rpc(P(41), wd11, 128, 256, 256);    rpc(P(43), wd12, 128, 128, 128);
    rpu(P(21), wu4, 2048, 1024); rpu(P(27), wu3, 1024, 512);
    rpu(P(33), wu2, 512, 256);   rpu(P(39), wu1, 256, 128);
    repack_head_w<<<(1232 * 128 + 255) / 256, 256, 0, stream>>>(P(45), whd);
    pack_noisy_kernel<<<(HW + 255) / 256, 256, 0, stream>>>(noisy, nz, H, W);

    // ---- launch helpers ---------------------------------------------------
    auto conv = [&](const _Float16* i1, const _Float16* i2, int c1, int c2,
                    int cpad, const _Float16* wg, const float* bs,
                    _Float16* o, int cout, int h, int w) {
        dim3 grid((h * w) / 64, cout / 128);
        conv3x3_wmma<<<grid, 256, 0, stream>>>(i1, i2, c1, c2, cpad, wg, bs,
                                               o, cout, h, w);
    };
    auto upcv = [&](const _Float16* i, const _Float16* wg, const float* bs,
                    _Float16* o, int cin, int cout, int h, int w) {
        dim3 grid((h * w) / 16, cout / 128, 4);
        upconv2x2_wmma<<<grid, 256, 0, stream>>>(i, wg, bs, o, cin, cout, h, w);
    };
    auto pool = [&](const _Float16* i, _Float16* o, int c, int h, int w) {
        int tot = c * (h / 2) * (w / 2);
        maxpool2_h<<<(tot + 255) / 256, 256, 0, stream>>>(i, o, c, h, w);
    };

    // ---- forward pass -----------------------------------------------------
    conv(nz, nz, 32, 32, 32,          we11, P(2),  t1, 128, 128, 128);
    conv(t1, t1, 128, 128, 128,       we12, P(4),  x1, 128, 128, 128);
    pool(x1, p1, 128, 128, 128);
    conv(p1, p1, 128, 128, 128,       we21, P(6),  t2, 256, 64, 64);
    conv(t2, t2, 256, 256, 256,       we22, P(8),  x2, 256, 64, 64);
    pool(x2, p2, 256, 64, 64);
    conv(p2, p2, 256, 256, 256,       we31, P(10), t3, 512, 32, 32);
    conv(t3, t3, 512, 512, 512,       we32, P(12), x3, 512, 32, 32);
    pool(x3, p3, 512, 32, 32);
    conv(p3, p3, 512, 512, 512,       we41, P(14), t4, 1024, 16, 16);
    conv(t4, t4, 1024, 1024, 1024,    we42, P(16), x4, 1024, 16, 16);
    pool(x4, p4, 1024, 16, 16);
    conv(p4, p4, 1024, 1024, 1024,    wbn1, P(18), bnm, 2048, 8, 8);
    conv(bnm, bnm, 2048, 2048, 2048,  wbn2, P(20), bno, 2048, 8, 8);

    upcv(bno, wu4, P(22), u4, 2048, 1024, 8, 8);                  // -> 16x16
    conv(u4, x4, 1024, 1024, 2048,    wd41, P(24), d4m, 1024, 16, 16);
    conv(d4m, d4m, 1024, 1024, 1024,  wd42, P(26), d4, 1024, 16, 16);
    upcv(d4, wu3, P(28), u3, 1024, 512, 16, 16);                  // -> 32x32
    conv(u3, x3, 512, 512, 1024,      wd31, P(30), d3m, 512, 32, 32);
    conv(d3m, d3m, 512, 512, 512,     wd32, P(32), d3, 512, 32, 32);
    upcv(d3, wu2, P(34), u2, 512, 256, 32, 32);                   // -> 64x64
    conv(u2, x2, 256, 256, 512,       wd21, P(36), d2m, 256, 64, 64);
    conv(d2m, d2m, 256, 256, 256,     wd22, P(38), d2, 256, 64, 64);
    upcv(d2, wu1, P(40), u1, 256, 128, 64, 64);                   // -> 128x128
    conv(u1, x1, 128, 128, 256,       wd11, P(42), d1m, 128, 128, 128);
    conv(d1m, d1m, 128, 128, 128,     wd12, P(44), d1, 128, 128, 128);

    // head + apply
    {
        const int Mtiles = (K2 + 15) / 16;           // 77
        dim3 grid(HW / 16, (Mtiles + 7) / 8);        // (1024, 10)
        kpn_head_wmma<<<grid, 256, 0, stream>>>(d1, whd, P(46), raw, K2, H, W);
    }
    kpn_apply<<<HW / 256, 256, 0, stream>>>(raw, noisy, (float*)d_out, H, W);
}
